// TopKTokenChoiceRouterLoRa_2302102471509
// MI455X (gfx1250) — compile-verified, matrix operand order checked by reading, not run on hardware
//
#include <hip/hip_runtime.h>
#include <stdint.h>

#define E_    8
#define H_    4096
#define L_    64
#define G_    4096
#define TOPK_ 2

#define MBLK   128   // tokens per block
#define KSTAGE 128   // K depth staged in LDS per iteration
#define BROW   18    // padded uint32 per (ks, n) row of staged B (conflict-free)
#define LROW   68    // padded float stride per logits row (conflict-free)

typedef __attribute__((ext_vector_type(16))) __bf16   v16bf;
typedef __attribute__((ext_vector_type(8)))  float    v8f;
typedef __attribute__((ext_vector_type(2)))  float    f32x2;
typedef __attribute__((ext_vector_type(4)))  float    f32x4;
typedef __attribute__((ext_vector_type(2)))  uint32_t u32x2;

__device__ __forceinline__ uint32_t pack_bf16(float a, float b) {
  union { __bf16 h[2]; uint32_t u; } r;
  r.h[0] = (__bf16)a;   // low 16 bits  = even k
  r.h[1] = (__bf16)b;   // high 16 bits = odd  k
  return r.u;
}

// A fragment (16x32 bf16): lanes 0-15 = M rows, K 0-7 & 16-23; lanes 16-31 same M, K 8-15 & 24-31.
// Per lane: contiguous float2 global loads (merge to b128; non-temporal: x lines are single-use).
__device__ __forceinline__ v16bf load_a_frag(const float* __restrict__ xrow, int hh, int kbase) {
  const f32x2* p0 = (const f32x2*)(xrow + hh + kbase);
  const f32x2* p1 = (const f32x2*)(xrow + hh + 16 + kbase);
  v16bf a;
#pragma unroll
  for (int j = 0; j < 4; ++j) {
    f32x2 f = __builtin_nontemporal_load(p0 + j);
    a[2*j]     = (__bf16)f.x;
    a[2*j + 1] = (__bf16)f.y;
  }
#pragma unroll
  for (int j = 0; j < 4; ++j) {
    f32x2 f = __builtin_nontemporal_load(p1 + j);
    a[8 + 2*j]     = (__bf16)f.x;
    a[8 + 2*j + 1] = (__bf16)f.y;
  }
  return a;
}

// B fragment (32x16 bf16, K x N): lane n holds column n; lanes 0-15 K 0-7/16-23, lanes 16-31 K 8-15/24-31.
// Staged LDS row = 16 packed-k dwords (pair p <-> k=2p,2p+1) padded to 18 dwords.
__device__ __forceinline__ v16bf load_b_frag(const uint32_t* ldsb, int ks, int n, int hi4) {
  const uint32_t* row = ldsb + (ks * 64 + n) * BROW;
  union { v16bf v; u32x2 q[4]; } b;
  b.q[0] = *(const u32x2*)(row + hi4);          // k pairs -> vgpr 0,1
  b.q[1] = *(const u32x2*)(row + hi4 + 2);      // vgpr 2,3
  b.q[2] = *(const u32x2*)(row + 8 + hi4);      // vgpr 4,5
  b.q[3] = *(const u32x2*)(row + 8 + hi4 + 2);  // vgpr 6,7
  return b.v;
}

__global__ __launch_bounds__(128) void router_lora_kernel(
    const float* __restrict__ x, const float* __restrict__ w1,
    float* __restrict__ outw, int* __restrict__ outi) {
  // max(B stage: 4*64*18*4 = 18432 B, logits: 128*68*4 = 34816 B)
  __shared__ __align__(16) uint8_t smem[MBLK * LROW * 4];
  uint32_t* ldsb = (uint32_t*)smem;
  float*    ldsl = (float*)smem;

  const int tid    = threadIdx.x;
  const int lane   = tid & 31;
  const int wave   = tid >> 5;
  const int lane15 = lane & 15;
  const int hi     = lane >> 4;
  const int kbase  = hi * 8;
  const int hi4    = hi * 4;

  const int bid  = blockIdx.x;
  const int e    = bid >> 5;   // / (G/MBLK)
  const int mblk = bid & 31;

  const float* w1e  = w1 + (size_t)e * (H_ * L_);
  const size_t tok0 = (size_t)e * G_ + (size_t)mblk * MBLK;
  const float* xrow0 = x + (tok0 + wave * 32 + lane15) * (size_t)H_;  // M-subtile 0
  const float* xrow1 = xrow0 + (size_t)16 * H_;                       // M-subtile 1

  const v8f vzero = {0.f, 0.f, 0.f, 0.f, 0.f, 0.f, 0.f, 0.f};
  v8f acc[2][4];
#pragma unroll
  for (int m = 0; m < 2; ++m)
#pragma unroll
    for (int nt = 0; nt < 4; ++nt) acc[m][nt] = vzero;

  for (int h0 = 0; h0 < H_; h0 += KSTAGE) {
    __syncthreads();
    // Cooperative stage of w1[e][h0:h0+128][0:64] as packed bf16.
    // Each thread handles 4 consecutive loras: two b128 loads (rows h, h+1; lanes 0-15
    // span a contiguous 256B run), 4 cvt_pk, 4 LDS stores at 72B stride (<=2-way conflict).
#pragma unroll
    for (int i = 0; i < (KSTAGE / 2) * (L_ / 4) / 128; ++i) {  // 8 iters
      int u    = tid + i * 128;        // unit = 4 packed dwords
      int n4   = (u & 15) * 4;
      int rest = u >> 4;
      int p    = rest & 15;
      int ks   = rest >> 4;
      int h    = h0 + ks * 32 + 2 * p;
      f32x4 r0 = *(const f32x4*)(w1e + (size_t)h * L_ + n4);        // even k row
      f32x4 r1 = *(const f32x4*)(w1e + (size_t)h * L_ + L_ + n4);   // odd  k row
#pragma unroll
      for (int c = 0; c < 4; ++c)
        ldsb[(ks * 64 + n4 + c) * BROW + p] = pack_bf16(r0[c], r1[c]);
    }
    __syncthreads();

#pragma unroll
    for (int ks = 0; ks < KSTAGE / 32; ++ks) {
      int hh = h0 + ks * 32;
      v16bf a0 = load_a_frag(xrow0, hh, kbase);
      v16bf a1 = load_a_frag(xrow1, hh, kbase);
#pragma unroll
      for (int nt = 0; nt < 4; ++nt) {
        v16bf bf = load_b_frag(ldsb, ks, nt * 16 + lane15, hi4);
        acc[0][nt] = __builtin_amdgcn_wmma_f32_16x16x32_bf16(
            false, a0, false, bf, (short)0, acc[0][nt], false, false);
        acc[1][nt] = __builtin_amdgcn_wmma_f32_16x16x32_bf16(
            false, a1, false, bf, (short)0, acc[1][nt], false, false);
      }
    }
  }

  __syncthreads();
  // Spill logits to LDS: C/D layout is vgpr r, lanes 0-15 -> M=r, lanes 16-31 -> M=r+8; N = 16*nt + lane15.
#pragma unroll
  for (int m = 0; m < 2; ++m)
#pragma unroll
    for (int nt = 0; nt < 4; ++nt)
#pragma unroll
      for (int r = 0; r < 8; ++r) {
        int row = wave * 32 + m * 16 + (hi ? r + 8 : r);
        int n   = nt * 16 + lane15;
        ldsl[row * LROW + n] = acc[m][nt][r];
      }
  __syncthreads();

  // One token per thread: top-2 scan (strict > keeps lowest index on ties, matching jax top_k)
  const float* rowp = ldsl + tid * LROW;
  float v1 = -3.402823466e38f, v2 = -3.402823466e38f;
  int i1 = 0, i2 = 0;
#pragma unroll
  for (int j = 0; j < L_; j += 4) {
    f32x4 q = *(const f32x4*)(rowp + j);
#pragma unroll
    for (int c = 0; c < 4; ++c) {
      float l = q[c];
      int   ix = j + c;
      if (l > v1)      { v2 = v1; i2 = i1; v1 = l; i1 = ix; }
      else if (l > v2) { v2 = l; i2 = ix; }
    }
  }
  float s = 0.f;
#pragma unroll
  for (int j = 0; j < L_; j += 4) {
    f32x4 q = *(const f32x4*)(rowp + j);
    s += __expf(q[0] - v1) + __expf(q[1] - v1) + __expf(q[2] - v1) + __expf(q[3] - v1);
  }
  float inv = 1.f / s;                 // softmax weight of top-1 (exp(0)/sum)
  size_t token = tok0 + tid;
  outw[token * 2 + 0] = inv;
  outw[token * 2 + 1] = __expf(v2 - v1) * inv;
  outi[token * 2 + 0] = i1;
  outi[token * 2 + 1] = i2;
}

extern "C" void kernel_launch(void* const* d_in, const int* in_sizes, int n_in,
                              void* d_out, int out_size, void* d_ws, size_t ws_size,
                              hipStream_t stream) {
  (void)in_sizes; (void)n_in; (void)out_size; (void)d_ws; (void)ws_size;
  const float* x  = (const float*)d_in[0];
  const float* w1 = (const float*)d_in[1];
  float* outw = (float*)d_out;
  int*   outi = (int*)d_out + (size_t)E_ * G_ * TOPK_;  // indices after 65536 fp32 weights
  dim3 grid(E_ * (G_ / MBLK));  // 256 blocks
  dim3 block(128);              // 4 waves
  hipLaunchKernelGGL(router_lora_kernel, grid, block, 0, stream, x, w1, outw, outi);
}
